// PrototypicalFastRCNNPredictor_36335423324183
// MI455X (gfx1250) — matrix operand order repaired and checked by reading, not compiled, not run on hardware
//
#include <hip/hip_runtime.h>
#include <hip/hip_bf16.h>

// ---------------------------------------------------------------------------
// PrototypicalFastRCNNPredictor — CDNA5 (gfx1250) bf16 WMMA implementation.
// All GEMMs run on v_wmma_f32_16x16x32_bf16 (wave32), double-buffered LDS
// staging via GLOBAL_LOAD_ASYNC_TO_LDS_B128 (ASYNCcnt) when available,
// with a software-pipelined global->VGPR->LDS fallback.
// ---------------------------------------------------------------------------

typedef __attribute__((ext_vector_type(16))) __bf16 v16bf;
typedef __attribute__((ext_vector_type(8)))  float  v8f;
typedef __attribute__((ext_vector_type(4)))  int    v4i;

// --- async global->LDS availability probe (device pass only) ----------------
#if defined(__HIP_DEVICE_COMPILE__) && defined(__gfx1250__) && defined(__has_builtin)
#if __has_builtin(__builtin_amdgcn_global_load_async_to_lds_b128)
#define ASYNC_COPY 1
#endif
#endif
#ifndef ASYNC_COPY
#define ASYNC_COPY 0
#endif

#if ASYNC_COPY
#define AS_GLOBAL __attribute__((address_space(1)))
#define AS_LDS    __attribute__((address_space(3)))

static __device__ __forceinline__ void stage_async16(const __bf16* g, __bf16* l) {
  // 16 bytes global -> LDS, tracked by ASYNCcnt
  __builtin_amdgcn_global_load_async_to_lds_b128((AS_GLOBAL v4i*)g, (AS_LDS v4i*)l,
                                                 0, 0);
}
static __device__ __forceinline__ void async_wait0() {
#if __has_builtin(__builtin_amdgcn_s_wait_asynccnt)
  __builtin_amdgcn_s_wait_asynccnt(0);
#else
  asm volatile("s_wait_asynccnt 0x0" ::: "memory");
#endif
}
#endif

static __device__ __forceinline__ v16bf pack_frag(uint4 lo, uint4 hi) {
  union { uint4 u[2]; v16bf v; } c;
  c.u[0] = lo; c.u[1] = hi;
  return c.v;
}

// ---------------------------------------------------------------------------
// fp32 -> bf16 elementwise convert (grid-stride)
// ---------------------------------------------------------------------------
__global__ void to_bf16_kernel(const float* __restrict__ in,
                               __bf16* __restrict__ out, size_t n) {
  size_t i = (size_t)blockIdx.x * blockDim.x + threadIdx.x;
  const size_t stride = (size_t)gridDim.x * blockDim.x;
  for (; i < n; i += stride) out[i] = (__bf16)in[i];
}

// ---------------------------------------------------------------------------
// W[K,N] fp32 (row-major)  ->  WT[N,K] bf16 (row-major), LDS-tiled transpose
// ---------------------------------------------------------------------------
__global__ __launch_bounds__(256)
void transpose_to_bf16_kernel(const float* __restrict__ W,
                              __bf16* __restrict__ WT, int K, int N) {
  __shared__ float tile[32][33];
  const int kb = blockIdx.y * 32, nb = blockIdx.x * 32;
  const int tx = threadIdx.x, ty = threadIdx.y;  // 32 x 8
#pragma unroll
  for (int i = 0; i < 32; i += 8) {
    const int k = kb + ty + i, n = nb + tx;
    tile[ty + i][tx] = (k < K && n < N) ? W[(size_t)k * N + n] : 0.0f;
  }
  __syncthreads();
#pragma unroll
  for (int i = 0; i < 32; i += 8) {
    const int n = nb + ty + i, k = kb + tx;
    if (n < N && k < K) WT[(size_t)n * K + k] = (__bf16)tile[tx][ty + i];
  }
}

// ---------------------------------------------------------------------------
// Tiled bf16 WMMA GEMM:  C[M,N] = epilogue(A[M,K] @ BT[N,K]^T + bias[N])
//   block tile 128x128, 8 waves, wave tile 32x64 (2x4 wmma frags), K step 32,
//   double-buffered LDS (ping-pong), LDS row stride 40 halves (80B) to spread
//   bank usage. Software pipeline: issue tile k+1, compute tile k, commit.
// ---------------------------------------------------------------------------
template <bool LRELU, bool OUT_BF16>
__global__ __launch_bounds__(256)
void gemm_bf16_wmma(const __bf16* __restrict__ A, const __bf16* __restrict__ BT,
                    const float* __restrict__ bias, void* __restrict__ Cout,
                    int Ktot, int Ntot, int ldc) {
  constexpr int LDST = 40;  // halves per LDS tile row
  __shared__ __bf16 As[2][128 * LDST];
  __shared__ __bf16 Bs[2][128 * LDST];

  const int tid  = threadIdx.x;
  const int bm   = blockIdx.y * 128;
  const int bn   = blockIdx.x * 128;
  const int wave = tid >> 5, lane = tid & 31;
  const int wm = wave & 3, wn = wave >> 2;     // 4x2 wave grid
  const int l16 = lane & 15, lh = lane >> 4;   // half-wave id

  // tile staging: 512 x 16B segments / 256 threads -> each thread owns
  // rows (lr, lr+64) at half-offset seg for both A and BT tiles.
  const int lr  = tid >> 2;        // 0..63
  const int seg = (tid & 3) * 8;   // 0,8,16,24 halves
  const __bf16* gA0 = A  + (size_t)(bm + lr)      * Ktot + seg;
  const __bf16* gA1 = A  + (size_t)(bm + lr + 64) * Ktot + seg;
  const __bf16* gB0 = BT + (size_t)(bn + lr)      * Ktot + seg;
  const __bf16* gB1 = BT + (size_t)(bn + lr + 64) * Ktot + seg;
  const int lOff0 = lr * LDST + seg;
  const int lOff1 = (lr + 64) * LDST + seg;

  const int nk = Ktot >> 5;
  v8f acc[2][4] = {};

  // ---- prologue: stage K-tile 0 into buffer 0 ----
#if ASYNC_COPY
  stage_async16(gA0, &As[0][lOff0]);
  stage_async16(gA1, &As[0][lOff1]);
  stage_async16(gB0, &Bs[0][lOff0]);
  stage_async16(gB1, &Bs[0][lOff1]);
  async_wait0();
#else
  {
    const uint4 a0 = *reinterpret_cast<const uint4*>(gA0);
    const uint4 a1 = *reinterpret_cast<const uint4*>(gA1);
    const uint4 b0 = *reinterpret_cast<const uint4*>(gB0);
    const uint4 b1 = *reinterpret_cast<const uint4*>(gB1);
    *reinterpret_cast<uint4*>(&As[0][lOff0]) = a0;
    *reinterpret_cast<uint4*>(&As[0][lOff1]) = a1;
    *reinterpret_cast<uint4*>(&Bs[0][lOff0]) = b0;
    *reinterpret_cast<uint4*>(&Bs[0][lOff1]) = b1;
  }
#endif
  __syncthreads();

  for (int kt = 0; kt < nk; ++kt) {
    const int cur = kt & 1, nxt = cur ^ 1;
    const int k1 = (kt + 1) << 5;
    const bool more = (kt + 1 < nk);

    // ---- issue next tile's staging (overlaps with WMMAs below) ----
#if ASYNC_COPY
    if (more) {
      stage_async16(gA0 + k1, &As[nxt][lOff0]);
      stage_async16(gA1 + k1, &As[nxt][lOff1]);
      stage_async16(gB0 + k1, &Bs[nxt][lOff0]);
      stage_async16(gB1 + k1, &Bs[nxt][lOff1]);
    }
#else
    uint4 na0, na1, nb0, nb1;
    if (more) {
      na0 = *reinterpret_cast<const uint4*>(gA0 + k1);
      na1 = *reinterpret_cast<const uint4*>(gA1 + k1);
      nb0 = *reinterpret_cast<const uint4*>(gB0 + k1);
      nb1 = *reinterpret_cast<const uint4*>(gB1 + k1);
    }
#endif

    // ---- fragments from current buffer ----
    // A frag: lanes 0-15 hold K=0..7 & 16..23, lanes 16-31 K=8..15 & 24..31
    v16bf afrag[2], bfrag[4];
#pragma unroll
    for (int mi = 0; mi < 2; ++mi) {
      const int r = wm * 32 + mi * 16 + l16;
      const uint4 lo = *reinterpret_cast<const uint4*>(&As[cur][r * LDST + lh * 8]);
      const uint4 hi = *reinterpret_cast<const uint4*>(&As[cur][r * LDST + 16 + lh * 8]);
      afrag[mi] = pack_frag(lo, hi);
    }
    // B frag: lane holds column l16, K=8*lh..8*lh+7 & 16+8*lh..16+8*lh+7
#pragma unroll
    for (int ni = 0; ni < 4; ++ni) {
      const int r = wn * 64 + ni * 16 + l16;
      const uint4 lo = *reinterpret_cast<const uint4*>(&Bs[cur][r * LDST + lh * 8]);
      const uint4 hi = *reinterpret_cast<const uint4*>(&Bs[cur][r * LDST + 16 + lh * 8]);
      bfrag[ni] = pack_frag(lo, hi);
    }

#pragma unroll
    for (int mi = 0; mi < 2; ++mi)
#pragma unroll
      for (int ni = 0; ni < 4; ++ni)
        acc[mi][ni] = __builtin_amdgcn_wmma_f32_16x16x32_bf16(
            false, afrag[mi], false, bfrag[ni], (short)0, acc[mi][ni], false, false);

    // ---- commit next tile ----
#if ASYNC_COPY
    if (more) async_wait0();
#else
    if (more) {
      *reinterpret_cast<uint4*>(&As[nxt][lOff0]) = na0;
      *reinterpret_cast<uint4*>(&As[nxt][lOff1]) = na1;
      *reinterpret_cast<uint4*>(&Bs[nxt][lOff0]) = nb0;
      *reinterpret_cast<uint4*>(&Bs[nxt][lOff1]) = nb1;
    }
#endif
    __syncthreads();
  }

  // ---- epilogue: C/D layout -> row = 8*lh + vgpr, col = l16 ----
#pragma unroll
  for (int ni = 0; ni < 4; ++ni) {
    const int col = bn + wn * 64 + ni * 16 + l16;
    if (col < Ntot) {
      const float b = bias[col];
#pragma unroll
      for (int mi = 0; mi < 2; ++mi) {
#pragma unroll
        for (int r = 0; r < 8; ++r) {
          const int row = bm + wm * 32 + mi * 16 + lh * 8 + r;
          float v = acc[mi][ni][r] + b;
          if (LRELU) v = (v > 0.0f) ? v : 0.01f * v;
          if (OUT_BF16)
            reinterpret_cast<__bf16*>(Cout)[(size_t)row * ldc + col] = (__bf16)v;
          else
            reinterpret_cast<float*>(Cout)[(size_t)row * ldc + col] = v;
        }
      }
    }
  }
}

// ---------------------------------------------------------------------------
// Row-wise L2 normalize: emb_pre[8192,256] fp32 -> emb fp32 (d_out) + bf16 (ws)
// ---------------------------------------------------------------------------
__global__ __launch_bounds__(256)
void normalize_emb_kernel(const float* __restrict__ epre, float* __restrict__ emb,
                          __bf16* __restrict__ embb) {
  const int row = blockIdx.x;
  const float v = epre[(size_t)row * 256 + threadIdx.x];
  __shared__ float red[8];
  float s = v * v;
#pragma unroll
  for (int off = 16; off; off >>= 1) s += __shfl_down(s, off, 32);
  if ((threadIdx.x & 31) == 0) red[threadIdx.x >> 5] = s;
  __syncthreads();
  if (threadIdx.x == 0) {
    float tot = 0.0f;
#pragma unroll
    for (int i = 0; i < 8; ++i) tot += red[i];
    red[0] = 1.0f / sqrtf(tot);
  }
  __syncthreads();
  const float o = v * red[0];
  emb[(size_t)row * 256 + threadIdx.x] = o;
  embb[(size_t)row * 256 + threadIdx.x] = (__bf16)o;
}

// ---------------------------------------------------------------------------
// scores[N,21]: d2 = 2 - 2*(e_hat . P_hat)  (both unit norm),
// proba = exp(-d2/(2 sigma^2)) * softmax(prior/max(prior))
// ---------------------------------------------------------------------------
__global__ __launch_bounds__(256)
void scores_kernel(const float* __restrict__ emb, const float* __restrict__ prototypes,
                   const float* __restrict__ sigmas, const float* __restrict__ prior,
                   float* __restrict__ scores) {
  __shared__ float Ps[20 * 256];
  __shared__ float invn[20], priors_s[20], inv2s2[20];
  __shared__ float prob_s[8][20];

  for (int i = threadIdx.x; i < 20 * 256; i += 256) Ps[i] = prototypes[i];
  __syncthreads();
  if (threadIdx.x < 20) {
    float ss = 0.0f;
    for (int d = 0; d < 256; ++d) { const float p = Ps[threadIdx.x * 256 + d]; ss += p * p; }
    invn[threadIdx.x] = 1.0f / sqrtf(ss);
    const float sg = sigmas[threadIdx.x];
    inv2s2[threadIdx.x] = 1.0f / (2.0f * sg * sg);
  }
  if (threadIdx.x == 0) {
    float mx = prior[0];
    for (int k = 1; k < 20; ++k) mx = fmaxf(mx, prior[k]);
    float e[20], sum = 0.0f;
    for (int k = 0; k < 20; ++k) { e[k] = __expf(prior[k] / mx); sum += e[k]; }
    for (int k = 0; k < 20; ++k) priors_s[k] = e[k] / sum;
  }
  __syncthreads();

  const int wave = threadIdx.x >> 5, lane = threadIdx.x & 31;
  const int row = blockIdx.x * 8 + wave;
  const float* er = emb + (size_t)row * 256;
  float e8[8];
#pragma unroll
  for (int d = 0; d < 8; ++d) e8[d] = er[lane * 8 + d];

  for (int k = 0; k < 20; ++k) {
    float part = 0.0f;
#pragma unroll
    for (int d = 0; d < 8; ++d) part += e8[d] * Ps[k * 256 + lane * 8 + d];
#pragma unroll
    for (int off = 16; off; off >>= 1) part += __shfl_xor(part, off, 32);
    if (lane == 0) {
      const float dot = part * invn[k];      // e_hat . P_hat
      const float d2 = 2.0f - 2.0f * dot;    // |e|=|P|=1
      prob_s[wave][k] = __expf(-d2 * inv2s2[k]) * priors_s[k];
    }
  }
  __syncthreads();

  const float p = (lane < 20) ? prob_s[wave][lane] : 0.0f;  // proba >= 0
  float m = p;
#pragma unroll
  for (int off = 16; off; off >>= 1) m = fmaxf(m, __shfl_xor(m, off, 32));
  float* srow = scores + (size_t)row * 21;
  if (lane == 0) srow[0] = 1.0f - m;
  if (lane < 20) srow[1 + lane] = p;
}

// ---------------------------------------------------------------------------
extern "C" void kernel_launch(void* const* d_in, const int* in_sizes, int n_in,
                              void* d_out, int out_size, void* d_ws, size_t ws_size,
                              hipStream_t stream) {
  constexpr int N = 8192, C = 1024, D = 256;
  constexpr int NB = 84;       // 4*(20+1)
  constexpr int H = 512, R = 12544;

  const float* x      = (const float*)d_in[0];
  const float* protos = (const float*)d_in[1];
  const float* sigmas = (const float*)d_in[2];
  const float* prior  = (const float*)d_in[3];
  const float* W1c = (const float*)d_in[4];  const float* b1c = (const float*)d_in[5];
  const float* W2c = (const float*)d_in[6];  const float* b2c = (const float*)d_in[7];
  const float* W1b = (const float*)d_in[8];  const float* b1b = (const float*)d_in[9];
  const float* W2b = (const float*)d_in[10]; const float* b2b = (const float*)d_in[11];
  const float* Wr1 = (const float*)d_in[12]; const float* br1 = (const float*)d_in[13];
  const float* Wr2 = (const float*)d_in[14]; const float* br2 = (const float*)d_in[15];

  // output layout: scores | bbox | emb | reconstructions (fp32, flat)
  float* out        = (float*)d_out;
  float* out_scores = out;
  float* out_bbox   = out + (size_t)N * 21;
  float* out_emb    = out + (size_t)N * (21 + NB);
  float* out_rec    = out + (size_t)N * (21 + NB + D);

  // workspace carve-out (256B aligned)
  char* W = (char*)d_ws;
  size_t off = 0;
  auto alloc = [&](size_t bytes) -> char* {
    char* p = W + off;
    off += (bytes + 255) & ~(size_t)255;
    return p;
  };
  __bf16* xb      = (__bf16*)alloc((size_t)N * C * 2);
  __bf16* W1cT    = (__bf16*)alloc((size_t)C * C * 2);
  __bf16* W2cT    = (__bf16*)alloc((size_t)D * C * 2);
  __bf16* W1bT    = (__bf16*)alloc((size_t)C * C * 2);
  __bf16* W2bT    = (__bf16*)alloc((size_t)128 * C * 2);  // padded to 128 rows
  __bf16* Wr1T    = (__bf16*)alloc((size_t)H * D * 2);
  __bf16* Wr2T    = (__bf16*)alloc((size_t)R * H * 2);
  __bf16* h1c     = (__bf16*)alloc((size_t)N * C * 2);
  __bf16* h1b     = (__bf16*)alloc((size_t)N * C * 2);
  float*  emb_pre = (float*) alloc((size_t)N * D * 4);
  __bf16* embb    = (__bf16*)alloc((size_t)N * D * 2);
  __bf16* hr      = (__bf16*)alloc((size_t)N * H * 2);
  (void)ws_size; (void)n_in; (void)in_sizes; (void)out_size;

  const dim3 tb(32, 8);
  auto tgrid = [](int K, int Nn) { return dim3((Nn + 31) / 32, (K + 31) / 32); };

  // 1) precision conversion / weight transposition
  to_bf16_kernel<<<4096, 256, 0, stream>>>(x, xb, (size_t)N * C);
  transpose_to_bf16_kernel<<<tgrid(C, C), tb, 0, stream>>>(W1c, W1cT, C, C);
  transpose_to_bf16_kernel<<<tgrid(C, D), tb, 0, stream>>>(W2c, W2cT, C, D);
  transpose_to_bf16_kernel<<<tgrid(C, C), tb, 0, stream>>>(W1b, W1bT, C, C);
  transpose_to_bf16_kernel<<<tgrid(C, NB), tb, 0, stream>>>(W2b, W2bT, C, NB);
  transpose_to_bf16_kernel<<<tgrid(D, H), tb, 0, stream>>>(Wr1, Wr1T, D, H);
  transpose_to_bf16_kernel<<<tgrid(H, R), tb, 0, stream>>>(Wr2, Wr2T, H, R);

  // 2) classification branch: h1c = lrelu(x@W1c+b1c); emb_pre = h1c@W2c+b2c
  gemm_bf16_wmma<true,  true ><<<dim3(C / 128, N / 128), 256, 0, stream>>>(xb,  W1cT, b1c, h1c,     C, C,  C);
  gemm_bf16_wmma<false, false><<<dim3(D / 128, N / 128), 256, 0, stream>>>(h1c, W2cT, b2c, emb_pre, C, D,  D);

  // 3) bbox branch
  gemm_bf16_wmma<true,  true ><<<dim3(C / 128, N / 128), 256, 0, stream>>>(xb,  W1bT, b1b, h1b,      C, C,  C);
  gemm_bf16_wmma<false, false><<<dim3(1,       N / 128), 256, 0, stream>>>(h1b, W2bT, b2b, out_bbox, C, NB, NB);

  // 4) normalize embeddings, then prototype scores
  normalize_emb_kernel<<<N, 256, 0, stream>>>(emb_pre, out_emb, embb);
  scores_kernel<<<N / 8, 256, 0, stream>>>(out_emb, protos, sigmas, prior, out_scores);

  // 5) reconstruction MLP (dominant GEMM: 8192x512x12544)
  gemm_bf16_wmma<true,  true ><<<dim3(H / 128, N / 128), 256, 0, stream>>>(embb, Wr1T, br1, hr,      D, H,  H);
  gemm_bf16_wmma<false, false><<<dim3(R / 128, N / 128), 256, 0, stream>>>(hr,   Wr2T, br2, out_rec, H, R,  R);
}
